// MaxPoolAggregator_60387240181928
// MI455X (gfx1250) — matrix operand order repaired and checked by reading, not compiled, not run on hardware
//
#include <hip/hip_runtime.h>
#include <hip/hip_bf16.h>

// ---------------------------------------------------------------------------
// GraphSAGE max-pool aggregator for MI455X (gfx1250, wave32, WMMA bf16)
//
//   agg  = segment_max(relu(X[edge_trg] @ fc_w + b), edge_src)  (agg aliases d_out)
//   out  = concat([X, agg]) @ W
//
// N=50000, E=800000, D=256 (K=512 for the final GEMM).
// Data path: X pre-converted to bf16 in d_ws; edge gather uses
// GLOBAL_LOAD_ASYNC_TO_LDS (ASYNCcnt); out-GEMM X-panel staged by the
// Tensor Data Mover (TENSORcnt); math is v_wmma_f32_16x16x32_bf16.
// Edge kernel uses 64-edge tiles: each B fragment feeds 2 WMMAs from registers.
// ---------------------------------------------------------------------------

#define N_NODES 50000
#define N_EDGES 800000
#define D 256

typedef __attribute__((ext_vector_type(16))) __bf16    v16bf;
typedef __attribute__((ext_vector_type(8)))  float     v8f;
typedef __attribute__((ext_vector_type(4)))  unsigned  u32x4;
typedef __attribute__((ext_vector_type(8)))  unsigned  u32x8;

union Frag {
    v16bf    v;
    unsigned u[8];
    uint4    q[2];
};

// f32 -> packed bf16x2, round-to-nearest-even (pure integer ops)
__device__ __forceinline__ unsigned pk_bf16(float a, float b) {
    unsigned ua = __float_as_uint(a);
    unsigned ub = __float_as_uint(b);
    ua += 0x7FFFu + ((ua >> 16) & 1u);
    ub += 0x7FFFu + ((ub >> 16) & 1u);
    return (ua >> 16) | (ub & 0xFFFF0000u);
}

__device__ __forceinline__ unsigned short cvt_bf16(float a) {
    unsigned ua = __float_as_uint(a);
    ua += 0x7FFFu + ((ua >> 16) & 1u);
    return (unsigned short)(ua >> 16);
}

// generic shared-pointer -> raw LDS byte offset (low 32 bits of the aperture addr)
__device__ __forceinline__ unsigned lds_off(const void* p) {
    return (unsigned)(uintptr_t)p;
}

// ---------------------------------------------------------------------------
// Kernel 0: zero the aggregation buffer (aliases d_out). 12500 blocks x 256.
// ---------------------------------------------------------------------------
__global__ void zero_agg_kernel(float4* __restrict__ agg4) {
    int i = blockIdx.x * blockDim.x + threadIdx.x;   // < N*D/4 = 3,200,000
    agg4[i] = make_float4(0.f, 0.f, 0.f, 0.f);
}

// ---------------------------------------------------------------------------
// Kernel 1: bf16 transposed weights in workspace.
//   fcwT[n][k] = bf16(fc_w[k][n])   (256 x 256)
//   wT  [n][k] = bf16(W[k][n])      (256 x 512)
// ---------------------------------------------------------------------------
__global__ void prep_weights_kernel(const float* __restrict__ fcw,
                                    const float* __restrict__ wm,
                                    unsigned short* __restrict__ fcwT,
                                    unsigned short* __restrict__ wT) {
    int i = blockIdx.x * blockDim.x + threadIdx.x;
    if (i < 256 * 256) {
        int k = i >> 8, n = i & 255;
        fcwT[(size_t)n * 256 + k] = cvt_bf16(fcw[i]);
    } else {
        int j = i - 256 * 256;                       // j < 512*256
        int k = j >> 8, n = j & 255;
        wT[(size_t)n * 512 + k] = cvt_bf16(wm[j]);
    }
}

// ---------------------------------------------------------------------------
// Kernel 2: X (f32, row-major) -> XB (bf16). One thread = 8 elements.
// N*D/8 = 1,600,000 threads -> 6250 blocks x 256.
// ---------------------------------------------------------------------------
__global__ void conv_x_kernel(const float* __restrict__ X,
                              unsigned short* __restrict__ XB) {
    int i = blockIdx.x * blockDim.x + threadIdx.x;
    const float4* sp = (const float4*)X + (size_t)i * 2;
    float4 a = sp[0], b = sp[1];
    ((uint4*)XB)[i] = make_uint4(pk_bf16(a.x, a.y), pk_bf16(a.z, a.w),
                                 pk_bf16(b.x, b.y), pk_bf16(b.z, b.w));
}

// ---------------------------------------------------------------------------
// Kernel 3: edge MLP (bf16 WMMA, f32 accum) + integer atomic-max scatter.
// One block = 64 edges, 12500 blocks x 256 threads (8 waves).
// Wave w: M-tiles {2*(w>>2), 2*(w>>2)+1}, N-tiles {(w&3)+4t | t=0..3};
// each B fragment is reused for 2 WMMAs (both M-tiles) from registers.
// ---------------------------------------------------------------------------
#define ASTRIDE_E 264   // 256 + 8 bf16 pad: 528 B row (16B-aligned, banks offset 4/row)

__global__ void edge_mlp_scatter_kernel(const unsigned short* __restrict__ XB,
                                        const int*   __restrict__ esrc,
                                        const int*   __restrict__ etrg,
                                        const unsigned short* __restrict__ fcwT,
                                        const float* __restrict__ fcb,
                                        float* __restrict__ agg) {
    __shared__ __align__(16) unsigned short As[64 * ASTRIDE_E];   // 64 edges x 256 K

    const int tileBase = blockIdx.x * 64;
    const int tid = threadIdx.x;

    // ---- stage: async gather, 64 rows x 512 B; 4 threads per row, 128 B each ----
    {
        int row = tid >> 2;          // 0..63
        int seg = tid & 3;           // 0..3, 128 B each
        int trg = etrg[tileBase + row];
        unsigned long long ga =
            (unsigned long long)(uintptr_t)(XB + (size_t)trg * D) + (unsigned)(seg * 128);
        unsigned la = lds_off(As + row * ASTRIDE_E) + (unsigned)(seg * 128);
        asm volatile(
            "global_load_async_to_lds_b128 %0, %1, off\n\t"
            "global_load_async_to_lds_b128 %0, %1, off offset:16\n\t"
            "global_load_async_to_lds_b128 %0, %1, off offset:32\n\t"
            "global_load_async_to_lds_b128 %0, %1, off offset:48\n\t"
            "global_load_async_to_lds_b128 %0, %1, off offset:64\n\t"
            "global_load_async_to_lds_b128 %0, %1, off offset:80\n\t"
            "global_load_async_to_lds_b128 %0, %1, off offset:96\n\t"
            "global_load_async_to_lds_b128 %0, %1, off offset:112"
            :: "v"(la), "v"(ga) : "memory");
    }
    asm volatile("s_wait_asynccnt 0x0" ::: "memory");
    __syncthreads();

    // ---- compute ----
    const int wv   = tid >> 5;
    const int lane = tid & 31;
    const int mtp  = wv >> 2;        // M-tile pair 0/1 -> tiles {2mtp, 2mtp+1}
    const int ng   = wv & 3;         // N-tile group
    const int m    = lane & 15;
    const int hi   = lane >> 4;

    v8f acc[2][4] = {};

    const unsigned* ar0 = (const unsigned*)(As + (mtp * 32 + m) * ASTRIDE_E);
    const unsigned* ar1 = (const unsigned*)(As + (mtp * 32 + 16 + m) * ASTRIDE_E);
#pragma unroll
    for (int kb = 0; kb < 8; ++kb) {
        const int kOff = kb * 32;
        const int base = kOff >> 1;

        Frag a0, a1;
#pragma unroll
        for (int j = 0; j < 8; ++j) {        // A 16x32: contiguous k = 16*hi + s, same order as B
            const int idx = base + j + 8 * hi;
            a0.u[j] = ar0[idx];
            a1.u[j] = ar1[idx];
        }

#pragma unroll
        for (int t = 0; t < 4; ++t) {
            const int nOff = (ng + 4 * t) * 16;
            Frag b;
            const uint4* bp = (const uint4*)(fcwT + (size_t)(nOff + m) * 256 + kOff + 16 * hi);
            b.q[0] = bp[0];
            b.q[1] = bp[1];
            acc[0][t] = __builtin_amdgcn_wmma_f32_16x16x32_bf16(
                false, a0.v, false, b.v, (short)0, acc[0][t], false, false);
            acc[1][t] = __builtin_amdgcn_wmma_f32_16x16x32_bf16(
                false, a1.v, false, b.v, (short)0, acc[1][t], false, false);
        }
    }

    // ---- scatter: bias add, int-bit-pattern atomic max (ReLU + zero-init fold) ----
#pragma unroll
    for (int t = 0; t < 4; ++t) {
        const int nOff = (ng + 4 * t) * 16;
        const int n = nOff + m;
        const float bn = fcb[n];
#pragma unroll
        for (int p = 0; p < 2; ++p) {
#pragma unroll
            for (int r = 0; r < 8; ++r) {
                const int M = r + 8 * hi;
                const int e = tileBase + (mtp * 2 + p) * 16 + M;
                const int src = esrc[e];
                const float v = acc[p][t][r] + bn;
                atomicMax((int*)(agg + (size_t)src * D + n), __float_as_int(v));
            }
        }
    }
}

// ---------------------------------------------------------------------------
// Kernel 4: out = concat([X, agg]) @ W, in place over the agg buffer.
// 1563 blocks x 256 threads, 32 nodes/block. X-panel (K 0..255) staged by the
// Tensor Data Mover with row padding; agg-panel (K 256..511) converted via VGPRs.
// ---------------------------------------------------------------------------
#define ASTRIDE_O 264   // per-panel row stride, matches TDM pad (512 B data + 16 B pad)

__global__ void out_gemm_kernel(const unsigned short* __restrict__ XB,
                                const unsigned short* __restrict__ wT,
                                float* __restrict__ outAgg) {
    __shared__ __align__(16) unsigned short AsX[32 * ASTRIDE_O];  // K 0..255 (bf16 of X)
    __shared__ __align__(16) unsigned short AsG[32 * ASTRIDE_O];  // K 256..511 (bf16 of agg)

    const int nodeBase = blockIdx.x * 32;
    const int tid = threadIdx.x;
    const int wv  = tid >> 5;

    // ---- TDM: DMA 32 x 512B tile of XB into AsX, 16B pad per row ----
    if (wv == 0) {
        unsigned long long ga =
            (unsigned long long)(uintptr_t)(XB + (size_t)nodeBase * D);
        unsigned remaining = (unsigned)(N_NODES - nodeBase);   // tail rows zero-filled
        u32x4 g0;
        g0[0] = 1u;                                            // count=1 (valid D#)
        g0[1] = lds_off(AsX);                                  // lds_addr
        g0[2] = (unsigned)ga;                                  // global_addr[31:0]
        g0[3] = (unsigned)((ga >> 32) & 0x01FFFFFFull) | (2u << 30);  // addr[56:32]|type=2
        u32x8 g1;
        g1[0] = (1u << 16)      // data_size = 2 bytes
              | (1u << 20)      // pad_enable
              | (6u << 22)      // pad_interval: every 128 dwords (512 B)
              | (3u << 25);     // pad_amount: 4 dwords (16 B)
        g1[1] = (256u & 0xFFFFu) << 16;            // tensor_dim0 lo16 (=256 elements)
        g1[2] = (remaining & 0xFFFFu) << 16;       // tensor_dim0 hi16=0 | tensor_dim1 lo16
        g1[3] = (remaining >> 16) | (256u << 16);  // tensor_dim1 hi16 | tile_dim0=256
        g1[4] = 32u;                               // tile_dim1=32, tile_dim2=0
        g1[5] = 256u;                              // tensor_dim0_stride lo32
        g1[6] = 0u;
        g1[7] = 0u;
        asm volatile("tensor_load_to_lds %0, %1" :: "s"(g0), "s"(g1) : "memory");
    }

    // ---- agg panel: load f32 rows of d_out, convert to bf16 into AsG ----
    {
        int row = tid >> 3;                  // 0..31
        int colBase = (tid & 7) * 32;        // 0..224 within the agg half
        int node = nodeBase + row;
        if (node >= N_NODES) node = N_NODES - 1;
        const float4* sp = (const float4*)(outAgg + (size_t)node * D + colBase);
        unsigned* dp = (unsigned*)(AsG + row * ASTRIDE_O + colBase);
#pragma unroll
        for (int i = 0; i < 8; ++i) {
            float4 f = sp[i];
            dp[2 * i + 0] = pk_bf16(f.x, f.y);
            dp[2 * i + 1] = pk_bf16(f.z, f.w);
        }
    }
    if (wv == 0) __builtin_amdgcn_s_wait_tensorcnt(0);
    __syncthreads();   // all agg reads + DMA complete before in-place overwrite

    const int lane = tid & 31;
    const int mt   = wv >> 2;
    const int ng   = wv & 3;
    const int m    = lane & 15;
    const int hi   = lane >> 4;
    const int arow = mt * 16 + m;

    v8f acc[4] = {};

    const unsigned* arX = (const unsigned*)(AsX + arow * ASTRIDE_O);
    const unsigned* arG = (const unsigned*)(AsG + arow * ASTRIDE_O);
#pragma unroll
    for (int kb = 0; kb < 16; ++kb) {
        const int kOff = kb * 32;
        const unsigned* ar = (kb < 8) ? arX : arG;
        const int base = ((kb < 8) ? kOff : (kOff - 256)) >> 1;

        Frag a;
#pragma unroll
        for (int j = 0; j < 8; ++j)
            a.u[j] = ar[base + j + 8 * hi];

#pragma unroll
        for (int t = 0; t < 4; ++t) {
            const int nOff = (ng + 4 * t) * 16;
            Frag b;
            const uint4* bp = (const uint4*)(wT + (size_t)(nOff + m) * 512 + kOff + 16 * hi);
            b.q[0] = bp[0];
            b.q[1] = bp[1];
            acc[t] = __builtin_amdgcn_wmma_f32_16x16x32_bf16(
                false, a.v, false, b.v, (short)0, acc[t], false, false);
        }
    }

    // ---- writeback (row tiles partitioned across blocks -> in-place safe) ----
#pragma unroll
    for (int t = 0; t < 4; ++t) {
        const int nOff = (ng + 4 * t) * 16;
        const int n = nOff + m;
#pragma unroll
        for (int r = 0; r < 8; ++r) {
            const int M = r + 8 * hi;
            const int node = nodeBase + mt * 16 + M;
            if (node < N_NODES)
                outAgg[(size_t)node * D + n] = acc[t][r];
        }
    }
}

// ---------------------------------------------------------------------------
extern "C" void kernel_launch(void* const* d_in, const int* in_sizes, int n_in,
                              void* d_out, int out_size, void* d_ws, size_t ws_size,
                              hipStream_t stream) {
    const float* X    = (const float*)d_in[0];           // [N, 256]
    const int*   adj  = (const int*)  d_in[1];           // [2, E]
    const float* fcw  = (const float*)d_in[2];           // [256, 256]
    const float* fcb  = (const float*)d_in[3];           // [256]
    const float* wm   = (const float*)d_in[4];           // [512, 256]
    float*       out  = (float*)d_out;                   // [N, 256], doubles as agg

    const int* esrc = adj;
    const int* etrg = adj + N_EDGES;

    unsigned short* fcwT = (unsigned short*)d_ws;        // 256*256 bf16 = 128 KB
    unsigned short* wT   = fcwT + 256 * 256;             // 256*512 bf16 = 256 KB
    unsigned short* XB   = wT + 256 * 512;               // N*256 bf16 = 25.6 MB

    // 0) zero aggregation buffer (3,200,000 float4 -> 12500 blocks)
    zero_agg_kernel<<<12500, 256, 0, stream>>>((float4*)out);

    // 1) bf16 transposed weights (196608 elements -> 768 blocks)
    prep_weights_kernel<<<768, 256, 0, stream>>>(fcw, wm, fcwT, wT);

    // 2) X -> bf16 (1,600,000 threads -> 6250 blocks)
    conv_x_kernel<<<6250, 256, 0, stream>>>(X, XB);

    // 3) edge MLP + scatter max (E/64 = 12500 edge tiles)
    edge_mlp_scatter_kernel<<<N_EDGES / 64, 256, 0, stream>>>(XB, esrc, etrg, fcwT, fcb, out);

    // 4) final GEMM, in place over the agg buffer (ceil(N/32) = 1563 node tiles)
    out_gemm_kernel<<<(N_NODES + 31) / 32, 256, 0, stream>>>(XB, wT, out);
}
